// LigerFusedLinearCosineSimilarityLoss_89240830476401
// MI455X (gfx1250) — compile-verified, matrix operand order checked by reading, not run on hardware
//
#include <hip/hip_runtime.h>
#include <hip/hip_bf16.h>

// ---------------------------------------------------------------------------
// Fused dual-GEMM (student & teacher logits) + CE/cosine distillation loss.
// MI455X / gfx1250: wave32, v_wmma_f32_16x16x32_bf16, f32 accumulate.
//
// Roofline: 537 GFLOP vs ~560 MB HBM floor (weights fp32) => compute bound on
// the WMMA pipes. Logits ([2048,32000] fp32, 262 MB each) never touch HBM;
// only ~10.3 MB of per-(row,colblock) reduction partials are written.
//
// Data path (CDNA5-specific): GLOBAL_LOAD_ASYNC_TO_LDS_B128 streams fp32
// tiles straight into LDS (no VGPR staging, ASYNCcnt tracked). fp32->bf16
// conversion at fragment build uses V_CVT_PK_BF16_F32 (1 VALU per pair, RNE),
// which co-executes with the XDL WMMAs. LDS rows are skewed by 16B (stride
// 144B = 36 banks) so 16-lane ds_load_b128 fragment reads are conflict-free.
// ---------------------------------------------------------------------------

typedef __attribute__((ext_vector_type(16))) __bf16 v16bf;
typedef __attribute__((ext_vector_type(8)))  float  v8f;

#define BT_DIM   2048
#define H_DIM    2048
#define V_DIM    32000
#define BM       128
#define BN       128
#define BK       32
#define NCB      (V_DIM / BN)    // 250 column blocks
#define KSTEPS   (H_DIM / BK)    // 64

#define ROW_BYTES   144          // 32 fp32 + 16B skew pad
#define TILE_BYTES  (128 * ROW_BYTES)       // 18432
#define STAGE_BYTES (4 * TILE_BYTES)        // 73728
#define LDS_BYTES   (2 * STAGE_BYTES)       // 147456 (dynamic; WGP has 320KB)

#define IGNORE_INDEX (-100)
#define W_HARD 0.5f
#define W_SOFT 0.5f
#define BETA_  0.5f
#define EPS_   1e-12f

union Frag { uint4 u[2]; v16bf v; };

__device__ __forceinline__ int   f2ord(float f) { int i = __float_as_int(f); return i >= 0 ? i : (i ^ 0x7FFFFFFF); }
__device__ __forceinline__ float ord2f(int i)   { return __int_as_float(i >= 0 ? i : (i ^ 0x7FFFFFFF)); }

// Two fp32 -> packed bf16 {hi[31:16], lo[15:0]}, RNE, single VALU op.
__device__ __forceinline__ unsigned pkbf(float lo, float hi) {
  unsigned r;
  asm("v_cvt_pk_bf16_f32 %0, %1, %2" : "=v"(r) : "v"(lo), "v"(hi));
  return r;
}

__device__ __forceinline__ void waitAsync0() {
  asm volatile("s_wait_asynccnt 0x0" ::: "memory");
}

// Issue one K-panel stage: 4 tiles x 128 rows x 32 fp32, via async DMA to LDS.
// Each thread issues 16 x b128 async copies; no VGPR data, no VALU convert.
__device__ __forceinline__ void issueStageAsync(const float* __restrict__ sIn,
                                                const float* __restrict__ tIn,
                                                const float* __restrict__ sW,
                                                const float* __restrict__ tW,
                                                int rowBase, int colBase, int k0,
                                                int tid, unsigned ldsStage) {
#pragma unroll
  for (int tt = 0; tt < 4; ++tt) {
    const float* base = (tt == 0) ? sIn : (tt == 1) ? tIn : (tt == 2) ? sW : tW;
    const int g0 = (tt < 2) ? rowBase : colBase;
#pragma unroll
    for (int ii = 0; ii < 4; ++ii) {
      const int idx  = tid + ii * 256;          // 0..1023
      const int row  = idx >> 3;                // 0..127
      const int chnk = idx & 7;                 // 8 x 16B per row
      const float* g = base + (size_t)(g0 + row) * H_DIM + k0 + chnk * 4;
      const unsigned lds = ldsStage + (unsigned)(tt * TILE_BYTES + row * ROW_BYTES + chnk * 16);
      asm volatile("global_load_async_to_lds_b128 %0, %1, off"
                   :: "v"(lds), "v"((unsigned long long)(size_t)g)
                   : "memory");
    }
  }
}

// Build a 16x32 bf16 fragment from an fp32 LDS tile (A layout; B identical
// with N playing the role of rows). Lane l<16: row rbase+l, K {0..7,16..23};
// lane>=16: row rbase+l-16, K {8..15,24..31}.
__device__ __forceinline__ Frag fragLoadCvt(const char* __restrict__ tile, int rbase, int lane) {
  const int row = rbase + (lane & 15);
  const int kb  = (lane >> 4) * 8;              // 0 or 8
  const char* p = tile + row * ROW_BYTES;
  const float4 a = *(const float4*)(p + (kb + 0)  * 4);
  const float4 b = *(const float4*)(p + (kb + 4)  * 4);
  const float4 c = *(const float4*)(p + (kb + 16) * 4);
  const float4 d = *(const float4*)(p + (kb + 20) * 4);
  Frag f;
  f.u[0].x = pkbf(a.x, a.y); f.u[0].y = pkbf(a.z, a.w);
  f.u[0].z = pkbf(b.x, b.y); f.u[0].w = pkbf(b.z, b.w);
  f.u[1].x = pkbf(c.x, c.y); f.u[1].y = pkbf(c.z, c.w);
  f.u[1].z = pkbf(d.x, d.y); f.u[1].w = pkbf(d.z, d.w);
  return f;
}

__global__ __launch_bounds__(256, 1)
void fused_dual_gemm_loss_partials(const float* __restrict__ sIn, const float* __restrict__ sW,
                                   const float* __restrict__ tIn, const float* __restrict__ tW,
                                   const int*   __restrict__ labels,
                                   const float* __restrict__ sBias, const float* __restrict__ tBias,
                                   float* __restrict__ pMax, float* __restrict__ pSum,
                                   float* __restrict__ pSt,  float* __restrict__ pSs,
                                   float* __restrict__ pTt,  float* __restrict__ labLogit) {
  const int tid  = threadIdx.x;
  const int lane = tid & 31;
  const int w    = tid >> 5;               // 8 waves
  const int wm   = w & 1;                  // 2 waves along M (64 rows each)
  const int wn   = w >> 1;                 // 4 waves along N (32 cols each)
  const int rowBase = blockIdx.x * BM;
  const int colBase = blockIdx.y * BN;
  const int nlo  = lane & 15;

  extern __shared__ __align__(16) char dsmem[];   // LDS_BYTES (144 KB)
  const unsigned ldsBase = (unsigned)(size_t)(void*)dsmem;  // flat addr low 32b == LDS offset

  v8f sAcc[4][2], tAcc[4][2];
#pragma unroll
  for (int mi = 0; mi < 4; ++mi)
#pragma unroll
    for (int ni = 0; ni < 2; ++ni) {
#pragma unroll
      for (int r = 0; r < 8; ++r) { sAcc[mi][ni][r] = 0.f; tAcc[mi][ni][r] = 0.f; }
    }

  // Prologue: stage 0 in flight, then wait + barrier.
  issueStageAsync(sIn, tIn, sW, tW, rowBase, colBase, 0, tid, ldsBase);
  waitAsync0();
  __syncthreads();

  for (int kk = 0; kk < KSTEPS; ++kk) {
    const int cur = kk & 1;
    if (kk + 1 < KSTEPS)                      // async copies for k+1 overlap the WMMAs below
      issueStageAsync(sIn, tIn, sW, tW, rowBase, colBase, (kk + 1) * BK,
                      tid, ldsBase + (unsigned)(((kk + 1) & 1) * STAGE_BYTES));
    if (kk + 2 < KSTEPS) {                    // L2 prefetch hints for the k+2 weight panel
      const float* pfs = sW + (size_t)(colBase + (tid >> 1)) * H_DIM + (kk + 2) * BK + (tid & 1) * 16;
      const float* pft = tW + (size_t)(colBase + (tid >> 1)) * H_DIM + (kk + 2) * BK + (tid & 1) * 16;
      __builtin_prefetch(pfs, 0, 1);
      __builtin_prefetch(pft, 0, 1);
    }

    const char* as  = dsmem + cur * STAGE_BYTES;
    const char* at  = as + TILE_BYTES;
    const char* wsl = as + 2 * TILE_BYTES;
    const char* wtl = as + 3 * TILE_BYTES;

    Frag aS[4], aT[4], bS[2], bT[2];
#pragma unroll
    for (int mi = 0; mi < 4; ++mi) {
      aS[mi] = fragLoadCvt(as, wm * 64 + mi * 16, lane);
      aT[mi] = fragLoadCvt(at, wm * 64 + mi * 16, lane);
    }
#pragma unroll
    for (int ni = 0; ni < 2; ++ni) {
      bS[ni] = fragLoadCvt(wsl, wn * 32 + ni * 16, lane);
      bT[ni] = fragLoadCvt(wtl, wn * 32 + ni * 16, lane);
    }
#pragma unroll
    for (int mi = 0; mi < 4; ++mi)
#pragma unroll
      for (int ni = 0; ni < 2; ++ni) {
        sAcc[mi][ni] = __builtin_amdgcn_wmma_f32_16x16x32_bf16(
            false, aS[mi].v, false, bS[ni].v, (short)0, sAcc[mi][ni], false, false);
        tAcc[mi][ni] = __builtin_amdgcn_wmma_f32_16x16x32_bf16(
            false, aT[mi].v, false, bT[ni].v, (short)0, tAcc[mi][ni], false, false);
      }

    waitAsync0();        // my stage-(k+1) copies landed in LDS
    __syncthreads();     // everyone's landed; safe to read next stage / overwrite prev
  }

  // ---------------- epilogue: bias + per-row block reductions ----------------
#pragma unroll
  for (int ni = 0; ni < 2; ++ni) {
    const int col = colBase + wn * 32 + ni * 16 + nlo;
    const float sb = sBias[col], tb = tBias[col];
#pragma unroll
    for (int mi = 0; mi < 4; ++mi)
#pragma unroll
      for (int r = 0; r < 8; ++r) { sAcc[mi][ni][r] += sb; tAcc[mi][ni][r] += tb; }
  }

  // Reuse LDS for the 5 x 128 reduction arrays.
  int*   blkMax = (int*)dsmem;
  float* blkSum = (float*)dsmem + 128;
  float* blkSt  = (float*)dsmem + 256;
  float* blkSs  = (float*)dsmem + 384;
  float* blkTt  = (float*)dsmem + 512;
  if (tid < 128) {
    blkMax[tid] = f2ord(-__builtin_inff());
    blkSum[tid] = 0.f; blkSt[tid] = 0.f; blkSs[tid] = 0.f; blkTt[tid] = 0.f;
  }
  __syncthreads();

  const bool leader = (nlo == 0);          // lanes 0 and 16
  // C layout: VGPR r / lane l holds row = wm*64+mi*16+(l>=16?8:0)+r,
  // col = colBase + wn*32 + ni*16 + (l&15).
#pragma unroll
  for (int mi = 0; mi < 4; ++mi)
#pragma unroll
    for (int r = 0; r < 8; ++r) {
      float s0 = sAcc[mi][0][r], s1 = sAcc[mi][1][r];
      float t0 = tAcc[mi][0][r], t1 = tAcc[mi][1][r];
      float pmax = fmaxf(s0, s1);
      float dst = s0 * t0 + s1 * t1;
      float dss = s0 * s0 + s1 * s1;
      float dtt = t0 * t0 + t1 * t1;
#pragma unroll
      for (int m = 1; m < 16; m <<= 1) {   // reduce over the 16 columns of the half-wave
        pmax = fmaxf(pmax, __shfl_xor(pmax, m, 32));
        dst += __shfl_xor(dst, m, 32);
        dss += __shfl_xor(dss, m, 32);
        dtt += __shfl_xor(dtt, m, 32);
      }
      if (leader) {
        const int lrow = wm * 64 + mi * 16 + (lane >> 4) * 8 + r;
        atomicMax(&blkMax[lrow], f2ord(pmax));
        atomicAdd(&blkSt[lrow], dst);
        atomicAdd(&blkSs[lrow], dss);
        atomicAdd(&blkTt[lrow], dtt);
      }
    }
  __syncthreads();

  // Phase B: sum of exp(s - rowmax), and the label logit (unique writer).
#pragma unroll
  for (int mi = 0; mi < 4; ++mi)
#pragma unroll
    for (int r = 0; r < 8; ++r) {
      const int lrow = wm * 64 + mi * 16 + (lane >> 4) * 8 + r;
      const float M = ord2f(blkMax[lrow]);
      const float s0 = sAcc[mi][0][r], s1 = sAcc[mi][1][r];
      float e = __expf(s0 - M) + __expf(s1 - M);
#pragma unroll
      for (int m = 1; m < 16; m <<= 1) e += __shfl_xor(e, m, 32);
      if (leader) atomicAdd(&blkSum[lrow], e);

      const int grow = rowBase + lrow;
      const int lbl  = labels[grow];
#pragma unroll
      for (int ni = 0; ni < 2; ++ni) {
        const int col = colBase + wn * 32 + ni * 16 + nlo;
        if (lbl == col) labLogit[grow] = (ni == 0 ? s0 : s1);
      }
    }
  __syncthreads();

  if (tid < 128) {
    const int grow = rowBase + tid;
    const size_t o = (size_t)grow * NCB + blockIdx.y;
    pMax[o] = ord2f(blkMax[tid]);
    pSum[o] = blkSum[tid];
    pSt[o]  = blkSt[tid];
    pSs[o]  = blkSs[tid];
    pTt[o]  = blkTt[tid];
  }
}

// -------- kernel 2: combine col-block partials per row, reduce to 3 scalars --------
__global__ __launch_bounds__(256)
void row_loss_reduce(const float* __restrict__ pMax, const float* __restrict__ pSum,
                     const float* __restrict__ pSt,  const float* __restrict__ pSs,
                     const float* __restrict__ pTt,  const float* __restrict__ labLogit,
                     const int* __restrict__ labels, float* __restrict__ acc) {
  const int tid = threadIdx.x;
  const int g = blockIdx.x * blockDim.x + tid;
  float hard = 0.f, soft = 0.f, nv = 0.f;
  if (g < BT_DIM) {
    const size_t base = (size_t)g * NCB;
    float M = -__builtin_inff();
    for (int b = 0; b < NCB; ++b) M = fmaxf(M, pMax[base + b]);
    float se = 0.f, dst = 0.f, dss = 0.f, dtt = 0.f;
    for (int b = 0; b < NCB; ++b) {
      se  += pSum[base + b] * __expf(pMax[base + b] - M);
      dst += pSt[base + b];
      dss += pSs[base + b];
      dtt += pTt[base + b];
    }
    const float lse = M + __logf(se);
    const int lbl = labels[g];
    const bool valid = (lbl != IGNORE_INDEX);
    hard = valid ? (lse - labLogit[g]) : 0.f;
    nv   = valid ? 1.f : 0.f;
    const float ns = fmaxf(sqrtf(dss), EPS_);
    const float nt = fmaxf(sqrtf(dtt), EPS_);
    soft = BETA_ * (1.f - dst / (ns * nt));
  }
  __shared__ float sh[3][256];
  sh[0][tid] = hard; sh[1][tid] = soft; sh[2][tid] = nv;
  __syncthreads();
  for (int s = 128; s > 0; s >>= 1) {
    if (tid < s) { sh[0][tid] += sh[0][tid + s]; sh[1][tid] += sh[1][tid + s]; sh[2][tid] += sh[2][tid + s]; }
    __syncthreads();
  }
  if (tid == 0) {
    atomicAdd(&acc[0], sh[0][0]);
    atomicAdd(&acc[1], sh[1][0]);
    atomicAdd(&acc[2], sh[2][0]);
  }
}

__global__ void init_acc_kernel(float* acc) {
  if (threadIdx.x < 3) acc[threadIdx.x] = 0.f;
}

__global__ void finalize_kernel(const float* acc, float* out) {
  if (threadIdx.x == 0) {
    const float nvalid = fmaxf(acc[2], 1.f);
    out[0] = W_HARD * (acc[0] / nvalid) + W_SOFT * (acc[1] / (float)BT_DIM);
  }
}

extern "C" void kernel_launch(void* const* d_in, const int* in_sizes, int n_in,
                              void* d_out, int out_size, void* d_ws, size_t ws_size,
                              hipStream_t stream) {
  (void)in_sizes; (void)n_in; (void)out_size; (void)ws_size;
  const float* sIn   = (const float*)d_in[0];
  const float* sW    = (const float*)d_in[1];
  const float* tIn   = (const float*)d_in[2];
  const float* tW    = (const float*)d_in[3];
  const int*   lbls  = (const int*)  d_in[4];
  const float* sBias = (const float*)d_in[5];
  const float* tBias = (const float*)d_in[6];

  float* w = (float*)d_ws;
  const size_t per = (size_t)BT_DIM * NCB;   // 512000
  float* pMax = w;
  float* pSum = w + per;
  float* pSt  = w + 2 * per;
  float* pSs  = w + 3 * per;
  float* pTt  = w + 4 * per;
  float* lab  = w + 5 * per;                 // BT_DIM
  float* acc  = lab + BT_DIM;                // 3 scalars

  init_acc_kernel<<<1, 32, 0, stream>>>(acc);

  dim3 grid(BT_DIM / BM, V_DIM / BN);        // 16 x 250; x-fast => weight-panel reuse in L2
  fused_dual_gemm_loss_partials<<<grid, 256, LDS_BYTES, stream>>>(
      sIn, sW, tIn, tW, lbls, sBias, tBias, pMax, pSum, pSt, pSs, pTt, lab);

  row_loss_reduce<<<BT_DIM / 256, 256, 0, stream>>>(pMax, pSum, pSt, pSs, pTt, lab, lbls, acc);

  finalize_kernel<<<1, 32, 0, stream>>>(acc, (float*)d_out);
}
